// DIN_54065048322145
// MI455X (gfx1250) — matrix operand order repaired and checked by reading, not compiled, not run on hardware
//
#include <hip/hip_runtime.h>
#include <stdint.h>

typedef __attribute__((ext_vector_type(16))) _Float16 v16h;
typedef __attribute__((ext_vector_type(8)))  float    v8f;

#define DIM    64
#define LHIST  200
#define LPAD   208          // 13 * 16
#define MTILES 13

// ---------------------------------------------------------------------------
// Weight pack: f32 [K][N] -> f16 fragments in WMMA B lane order.
//   dst[((kk*NT + nt)*32 + lane)*16 + i] = W[kk*32 + (i>>3)*16 + (lane>>4)*8 + (i&7)][nt*16 + (lane&15)]
// One contiguous 32B load per B-fragment afterwards.
// ---------------------------------------------------------------------------
__device__ inline void pack_one(_Float16* __restrict__ dst,
                                const float* __restrict__ src,
                                int KK, int NT, int N, int gid, int gstride) {
  const int total = KK * NT * 32 * 16;
  for (int i = gid; i < total; i += gstride) {
    int idx = i;
    const int e    = idx & 15;  idx >>= 4;
    const int lane = idx & 31;  idx >>= 5;
    const int nt   = idx % NT;
    const int kk   = idx / NT;
    const int col  = nt * 16 + (lane & 15);
    const int k    = kk * 32 + (e >> 3) * 16 + (lane >> 4) * 8 + (e & 7);
    dst[i] = (_Float16)src[(size_t)k * N + col];
  }
}

__global__ __launch_bounds__(256) void pack_weights_kernel(
    const float* __restrict__ aW1,   // [192][64]
    const float* __restrict__ fW1,   // [192][128]
    const float* __restrict__ fW2,   // [128][64]
    _Float16* __restrict__ pA,
    _Float16* __restrict__ pF1,
    _Float16* __restrict__ pF2) {
  const int gid = blockIdx.x * blockDim.x + threadIdx.x;
  const int gstride = gridDim.x * blockDim.x;
  pack_one(pA,  aW1, 6, 4, 64,  gid, gstride);
  pack_one(pF1, fW1, 6, 8, 128, gid, gstride);
  pack_one(pF2, fW2, 4, 4, 64,  gid, gstride);
}

// ---------------------------------------------------------------------------
// Kernel A: per-batch attention.  One block (8 wave32) per batch element.
//   s_stage: f32 hist during gather/prep, then reused as the scaled-relu
//            h matrix during the GEMM (aliased -> no LDS growth).
// ---------------------------------------------------------------------------
__global__ __launch_bounds__(256) void din_attn_kernel(
    const int*      __restrict__ item_ids,
    const int*      __restrict__ history,
    const int*      __restrict__ hist_len,
    const float*    __restrict__ emb_table,   // [V][64]
    const _Float16* __restrict__ packA,       // packed aW1 fragments
    const float*    __restrict__ ab1,         // [64]
    const float*    __restrict__ aW2,         // [64]
    const float*    __restrict__ ab2,         // [1]
    _Float16*       __restrict__ comb_out)    // [B][192] f16
{
  __shared__ float    s_stage[LPAD * DIM];    // 52 KB: hist f32, then h*w2
  __shared__ _Float16 s_f16[LPAD * 128];      // 52 KB: [hist_f16 | (hist*tgt)_f16]
  __shared__ float    s_tgt[DIM];
  __shared__ float    s_w2[DIM];
  __shared__ float    s_b1[DIM];
  __shared__ float    s_wgt[LPAD];
  __shared__ float    s_red[256];
  __shared__ float    s_pp[4][DIM];

  const int b    = blockIdx.x;
  const int tid  = threadIdx.x;
  const int lane = tid & 31;
  const int wave = tid >> 5;
  const int half = lane >> 4;      // 0 | 1
  const int l16  = lane & 15;
  const int hlen = hist_len[b];

  if (tid < DIM) {
    s_tgt[tid] = emb_table[(size_t)item_ids[b] * DIM + tid];
    s_w2[tid]  = aW2[tid];
    s_b1[tid]  = ab1[tid];
  }
  if (tid >= LHIST && tid < LPAD) s_wgt[tid] = 0.0f;

  // zero masked + padded rows [hlen, LPAD)
  for (int i = tid; i < (LPAD - hlen) * DIM; i += 256)
    s_stage[hlen * DIM + i] = 0.0f;

  // async gather rows [0, hlen): each lane moves one 16B segment (L2-resident)
  {
    const uint32_t lds_base = (uint32_t)(uintptr_t)(&s_stage[0]);
    const int nchunks = hlen * 16;             // 16 segments per 256B row
    for (int c = tid; c < nchunks; c += 256) {
      const int row = c >> 4, seg = c & 15;
      const int idx = history[(size_t)b * LHIST + row];
      const float* src = emb_table + (size_t)idx * DIM + seg * 4;
      const uint32_t dst = lds_base + (uint32_t)(row * DIM + seg * 4) * 4u;
      asm volatile("global_load_async_to_lds_b128 %0, %1, off"
                   :: "v"(dst), "v"(src) : "memory");
    }
    asm volatile("s_wait_asynccnt 0" ::: "memory");
  }
  __syncthreads();

  // ---- prep: f16 hist + f16 hist*tgt (one pass, reused by all 4 N-tiles) ----
  for (int i = tid; i < LPAD * DIM; i += 256) {
    const int row = i >> 6, d = i & 63;
    const float h = s_stage[i];
    s_f16[row * 128 + d]      = (_Float16)h;
    s_f16[row * 128 + 64 + d] = (_Float16)(h * s_tgt[d]);
  }
  __syncthreads();   // s_stage (f32 hist) is dead past here -> becomes h

  // target-only A fragments (K regions 64..127): row-invariant, built once
  v16h atgt[2];
  #pragma unroll
  for (int q = 0; q < 2; ++q) {
    #pragma unroll
    for (int j = 0; j < 8; ++j) {
      const int ka = q * 32 + half * 8 + j;
      atgt[q][j]     = (_Float16)s_tgt[ka];
      atgt[q][j + 8] = (_Float16)s_tgt[ka + 16];
    }
  }

  // ---- attention GEMM: [LPAD x 192] x [192 x 64] ----
  // epilogue: store relu(h+b1)*aW2[col] to s_stage; no cross-lane reduction.
  for (int nt = 0; nt < 4; ++nt) {
    const int col = nt * 16 + l16;
    v16h bf[6];
    #pragma unroll
    for (int kk = 0; kk < 6; ++kk)
      bf[kk] = *(const v16h*)(packA + ((size_t)(kk * 4 + nt) * 32 + lane) * 16);
    const float w2c = s_w2[col];
    const float b1c = s_b1[col];

    for (int mt = wave; mt < MTILES; mt += 8) {
      const int rowA = mt * 16 + l16;
      v8f acc = {};
      // K 0..63: hist
      #pragma unroll
      for (int kk = 0; kk < 2; ++kk) {
        const _Float16* p = &s_f16[rowA * 128 + kk * 32 + half * 8];
        v16h af;
        #pragma unroll
        for (int j = 0; j < 8; ++j) { af[j] = p[j]; af[j + 8] = p[16 + j]; }
        acc = __builtin_amdgcn_wmma_f32_16x16x32_f16(
            false, af, false, bf[kk], (short)0, acc, false, false);
      }
      // K 64..127: target (register fragments)
      acc = __builtin_amdgcn_wmma_f32_16x16x32_f16(
          false, atgt[0], false, bf[2], (short)0, acc, false, false);
      acc = __builtin_amdgcn_wmma_f32_16x16x32_f16(
          false, atgt[1], false, bf[3], (short)0, acc, false, false);
      // K 128..191: hist*target
      #pragma unroll
      for (int kk = 4; kk < 6; ++kk) {
        const _Float16* p = &s_f16[rowA * 128 + 64 + (kk - 4) * 32 + half * 8];
        v16h af;
        #pragma unroll
        for (int j = 0; j < 8; ++j) { af[j] = p[j]; af[j + 8] = p[16 + j]; }
        acc = __builtin_amdgcn_wmma_f32_16x16x32_f16(
            false, af, false, bf[kk], (short)0, acc, false, false);
      }
      #pragma unroll
      for (int r = 0; r < 8; ++r) {
        const float v = fmaxf(acc[r] + b1c, 0.0f) * w2c;
        s_stage[(mt * 16 + half * 8 + r) * DIM + col] = v;
      }
    }
  }
  __syncthreads();

  // ---- scores: row-sum of s_stage, then softmax over [0, 200) ----
  float sc = -3.0e38f;
  if (tid < LHIST) {
    const float4* p = (const float4*)&s_stage[tid * DIM];
    float s0 = 0.0f, s1 = 0.0f;
    #pragma unroll
    for (int i = 0; i < 16; i += 2) {
      const float4 a = p[i], c = p[i + 1];
      s0 += a.x + a.y + a.z + a.w;
      s1 += c.x + c.y + c.z + c.w;
    }
    sc = s0 + s1 + ab2[0];
  }
  s_red[tid] = sc;
  __syncthreads();
  for (int s = 128; s > 0; s >>= 1) {
    if (tid < s) s_red[tid] = fmaxf(s_red[tid], s_red[tid + s]);
    __syncthreads();
  }
  const float mx = s_red[0];
  __syncthreads();
  float e = 0.0f;
  if (tid < LHIST) e = __expf(sc - mx);
  s_red[tid] = e;
  __syncthreads();
  for (int s = 128; s > 0; s >>= 1) {
    if (tid < s) s_red[tid] += s_red[tid + s];
    __syncthreads();
  }
  const float inv = 1.0f / s_red[0];
  if (tid < LHIST) s_wgt[tid] = e * inv;
  __syncthreads();

  // ---- pooled = w^T hist (4-way L split, hist from f16 copy) ----
  {
    const int dcol  = tid & 63;
    const int chunk = tid >> 6;                 // 0..3, 50 rows each
    float acc = 0.0f;
    for (int l = chunk * 50; l < chunk * 50 + 50; ++l)
      acc += s_wgt[l] * (float)s_f16[l * 128 + dcol];
    s_pp[chunk][dcol] = acc;
  }
  __syncthreads();

  if (tid < DIM) {
    const float pooled = s_pp[0][tid] + s_pp[1][tid] + s_pp[2][tid] + s_pp[3][tid];
    const float tg = s_tgt[tid];
    _Float16* o = comb_out + (size_t)b * 192;
    o[tid]       = (_Float16)tg;
    o[64 + tid]  = (_Float16)pooled;
    o[128 + tid] = (_Float16)(pooled - tg);
  }
}

// ---------------------------------------------------------------------------
// Kernel B: final MLP, one wave per 16 batch rows.
// ---------------------------------------------------------------------------
__global__ __launch_bounds__(32) void din_mlp_kernel(
    const _Float16* __restrict__ comb,   // [B][192] f16
    const _Float16* __restrict__ pF1, const float* __restrict__ fb1,
    const _Float16* __restrict__ pF2, const float* __restrict__ fb2,
    const float* __restrict__ fW3, const float* __restrict__ fb3,
    float* __restrict__ out)             // [B]
{
  __shared__ _Float16 s_z1[16 * 128];
  __shared__ float    s_z2[16 * 64];

  const int mbase = blockIdx.x * 16;
  const int lane  = threadIdx.x;
  const int half  = lane >> 4;
  const int l16   = lane & 15;

  // ---- layer 1: z1 = relu(comb @ fW1 + fb1) ----
  for (int nt = 0; nt < 8; ++nt) {
    const int col = nt * 16 + l16;
    v8f acc = {};
    #pragma unroll
    for (int kk = 0; kk < 6; ++kk) {
      v16h af;
      const _Float16* arow =
          comb + (size_t)(mbase + l16) * 192 + kk * 32 + half * 8;
      #pragma unroll
      for (int j = 0; j < 8; ++j) { af[j] = arow[j]; af[j + 8] = arow[16 + j]; }
      const v16h bf = *(const v16h*)(pF1 + ((size_t)(kk * 8 + nt) * 32 + lane) * 16);
      acc = __builtin_amdgcn_wmma_f32_16x16x32_f16(
          false, af, false, bf, (short)0, acc, false, false);
    }
    const float b1 = fb1[col];
    #pragma unroll
    for (int r = 0; r < 8; ++r)
      s_z1[(half * 8 + r) * 128 + col] = (_Float16)fmaxf(acc[r] + b1, 0.0f);
  }
  __syncthreads();

  // ---- layer 2: z2 = relu(z1 @ fW2 + fb2) * fW3 (pre-scaled, stored) ----
  for (int nt = 0; nt < 4; ++nt) {
    const int col = nt * 16 + l16;
    v8f acc = {};
    #pragma unroll
    for (int kk = 0; kk < 4; ++kk) {
      v16h af;
      const _Float16* arow = s_z1 + (size_t)l16 * 128 + kk * 32 + half * 8;
      #pragma unroll
      for (int j = 0; j < 8; ++j) { af[j] = arow[j]; af[j + 8] = arow[16 + j]; }
      const v16h bf = *(const v16h*)(pF2 + ((size_t)(kk * 4 + nt) * 32 + lane) * 16);
      acc = __builtin_amdgcn_wmma_f32_16x16x32_f16(
          false, af, false, bf, (short)0, acc, false, false);
    }
    const float b2 = fb2[col];
    const float w3 = fW3[col];
    #pragma unroll
    for (int r = 0; r < 8; ++r)
      s_z2[(half * 8 + r) * 64 + col] = fmaxf(acc[r] + b2, 0.0f) * w3;
  }
  __syncthreads();

  // ---- layer 3: out = sigmoid(rowsum(z2) + fb3) ----
  if (lane < 16) {
    const float4* p = (const float4*)&s_z2[lane * 64];
    float s0 = 0.0f, s1 = 0.0f;
    #pragma unroll
    for (int i = 0; i < 16; i += 2) {
      const float4 a = p[i], c = p[i + 1];
      s0 += a.x + a.y + a.z + a.w;
      s1 += c.x + c.y + c.z + c.w;
    }
    const float z = s0 + s1 + fb3[0];
    out[mbase + lane] = 1.0f / (1.0f + __expf(-z));
  }
}

// ---------------------------------------------------------------------------
extern "C" void kernel_launch(void* const* d_in, const int* in_sizes, int n_in,
                              void* d_out, int out_size, void* d_ws, size_t ws_size,
                              hipStream_t stream) {
  const int*   item_ids = (const int*)d_in[0];
  const int*   history  = (const int*)d_in[1];
  const int*   hist_len = (const int*)d_in[2];
  const float* emb      = (const float*)d_in[3];
  const float* aW1      = (const float*)d_in[4];
  const float* ab1      = (const float*)d_in[5];
  const float* aW2      = (const float*)d_in[6];
  const float* ab2      = (const float*)d_in[7];
  const float* fW1      = (const float*)d_in[8];
  const float* fb1      = (const float*)d_in[9];
  const float* fW2      = (const float*)d_in[10];
  const float* fb2      = (const float*)d_in[11];
  const float* fW3      = (const float*)d_in[12];
  const float* fb3      = (const float*)d_in[13];

  const int B = in_sizes[0];                     // 4096

  // workspace layout (halves): comb | packA | packF1 | packF2
  _Float16* comb = (_Float16*)d_ws;              // B*192
  _Float16* pA   = comb + (size_t)B * 192;       // 6*4*32*16  = 12288
  _Float16* pF1  = pA   + 6 * 4 * 32 * 16;       // 6*8*32*16  = 24576
  _Float16* pF2  = pF1  + 6 * 8 * 32 * 16;       // 4*4*32*16  = 8192

  float* out = (float*)d_out;

  pack_weights_kernel<<<64, 256, 0, stream>>>(aW1, fW1, fW2, pA, pF1, pF2);
  din_attn_kernel<<<B, 256, 0, stream>>>(item_ids, history, hist_len, emb,
                                         pA, ab1, aW2, ab2, comb);
  din_mlp_kernel<<<B / 16, 32, 0, stream>>>(comb, pF1, fb1, pF2, fb2, fW3, fb3,
                                            out);
}
